// AttentionBlock_9251359556097
// MI455X (gfx1250) — matrix operand order repaired and checked by reading, not compiled, not run on hardware
//
#include <hip/hip_runtime.h>

#define DIMC   512
#define HEADS  8
#define HD     64
#define NGRP   32
#define BATCH  8
#define HW     1024
#define LCTX   77
#define CTXD   768
#define KVLEN  1101   /* 77 + 1024 */
#define KVPAD  1120   /* 35 * 32   */
#define EPSV   1e-5f
#define SCALE2 0.125f /* (HD^-0.25)^2 = 64^-0.5 */

typedef __bf16 bf16;
typedef __attribute__((ext_vector_type(16))) __bf16 v16bf;
typedef __attribute__((ext_vector_type(8)))  float  v8f;
typedef __attribute__((ext_vector_type(4)))  unsigned int u32x4;
typedef __attribute__((ext_vector_type(8)))  int          i32x8;
typedef __attribute__((ext_vector_type(4)))  int          i32x4;

// ---------------------------------------------------------------------------
// WMMA helper (CDNA5: D = A(16x32 bf16) * B(32x16 bf16) + C(16x16 f32))
// ---------------------------------------------------------------------------
__device__ __forceinline__ v8f wmma_bf16(v16bf a, v16bf b, v8f c) {
  return __builtin_amdgcn_wmma_f32_16x16x32_bf16(false, a, false, b,
                                                 (short)0, c, false, false);
}

// A fragment: row-major A[M x K]; lane m = lane&15; elem e -> k = (lane>>4)*16 + e
__device__ __forceinline__ v16bf load_a_frag(const bf16* __restrict__ A, int mbase,
                                             int ldk, int kk, int lane) {
  int half = lane >> 4, m = lane & 15;
  const bf16* p = A + (size_t)(mbase + m) * ldk + kk + (half << 4);
  v16bf a;
#pragma unroll
  for (int e = 0; e < 8; ++e) a[e] = p[e];          // k = half*16 + e
#pragma unroll
  for (int e = 0; e < 8; ++e) a[8 + e] = p[8 + e]; // k = half*16 + 8 + e
  return a;
}

// B fragment from N-major global storage: lane n = lane&15; elem e -> k = (lane>>4)*16 + e
__device__ __forceinline__ v16bf load_b_frag(const bf16* __restrict__ Bm, int nbase,
                                             int ldk, int kk, int lane) {
  int half = lane >> 4, n = lane & 15;
  const bf16* p = Bm + (size_t)(nbase + n) * ldk + kk + (half << 4);
  v16bf b;
#pragma unroll
  for (int e = 0; e < 16; ++e) b[e] = p[e];
  return b;
}

// ---------------------------------------------------------------------------
// Tensor Data Mover: DMA a (rows x cols) bf16 tile (row stride in elements)
// from global into LDS at lds_byte_off.  D# per CDNA5 ISA ch.8.
// (6-arg builtin form: groups 0..3 + extra group + cache policy)
// ---------------------------------------------------------------------------
__device__ __forceinline__ void tdm_load_panel(unsigned lds_byte_off,
                                               const bf16* gsrc,
                                               int rows, int cols,
                                               int row_stride_elems) {
  unsigned long long ga = (unsigned long long)(size_t)gsrc;
  u32x4 g0;
  g0[0] = 1u;                                        // count=1 (valid user D#)
  g0[1] = lds_byte_off;                              // lds_addr [63:32]
  g0[2] = (unsigned)(ga & 0xffffffffu);              // global_addr [95:64]
  g0[3] = (unsigned)((ga >> 32) & 0x01ffffffu)       // global_addr [120:96]
        | 0x80000000u;                               // type=2 ("image") [127:126]
  i32x8 g1;
  g1[0] = 0x00010000;                                // data_size=1 (2 bytes)
  g1[1] = (cols & 0xffff) << 16;                     // tensor_dim0[15:0]
  g1[2] = ((cols >> 16) & 0xffff)                    // tensor_dim0[31:16]
        | ((rows & 0xffff) << 16);                   // tensor_dim1[15:0]
  g1[3] = ((rows >> 16) & 0xffff)                    // tensor_dim1[31:16]
        | ((cols & 0xffff) << 16);                   // tile_dim0
  g1[4] = rows & 0xffff;                             // tile_dim1 (tile_dim2=0)
  g1[5] = row_stride_elems;                          // tensor_dim0_stride[31:0]
  g1[6] = 0;                                         // stride0 hi / stride1 lo
  g1[7] = 0;
  i32x4 gz4 = {0, 0, 0, 0};                          // 2D tensor: groups 2/3 unused
  i32x8 gz8 = {0, 0, 0, 0, 0, 0, 0, 0};
  __builtin_amdgcn_tensor_load_to_lds(g0, g1, gz4, gz4, gz8, 0);
}

// ---------------------------------------------------------------------------
// fp32 -> bf16 conversion
// ---------------------------------------------------------------------------
__global__ void f2bf_kernel(const float* __restrict__ in, bf16* __restrict__ out, int n) {
  int i = blockIdx.x * blockDim.x + threadIdx.x;
  if (i < n) out[i] = (bf16)in[i];
}

// ---------------------------------------------------------------------------
// GroupNorm:  x (B, C, HW) -> xnT (B, HW, C) bf16, gamma/beta applied
// ---------------------------------------------------------------------------
__global__ void groupnorm_kernel(const float* __restrict__ x,
                                 const float* __restrict__ gamma,
                                 const float* __restrict__ beta,
                                 bf16* __restrict__ xnT) {
  const int CPG = DIMC / NGRP;      // 16
  const int N   = CPG * HW;         // 16384
  int b = blockIdx.x >> 5;
  int g = blockIdx.x & 31;
  const float* xp = x + ((size_t)b * DIMC + g * CPG) * HW;
  int tid = threadIdx.x;

  float s = 0.f, ss = 0.f;
  for (int i = tid; i < N; i += 256) { float v = xp[i]; s += v; ss += v * v; }
  __shared__ float rs[256], rss[256];
  rs[tid] = s; rss[tid] = ss;
  __syncthreads();
  for (int off = 128; off > 0; off >>= 1) {
    if (tid < off) { rs[tid] += rs[tid + off]; rss[tid] += rss[tid + off]; }
    __syncthreads();
  }
  float mu   = rs[0] / (float)N;
  float var  = rss[0] / (float)N - mu * mu;
  float rinv = rsqrtf(var + EPSV);

  for (int i = tid; i < N; i += 256) {
    int cc = i >> 10;
    int hw = i & 1023;
    int c  = g * CPG + cc;
    float y = (xp[i] - mu) * rinv * gamma[c] + beta[c];
    xnT[((size_t)b * HW + hw) * DIMC + c] = (bf16)y;
  }
}

// ---------------------------------------------------------------------------
// Shared GEMM mainloop: workgroup = 4 waves, output block 64(M) x 64(N).
// B panel (64 rows x 128-col K slab) staged to LDS by the TDM, K = 512.
// Each wave computes a 16x64 tile: 1 A frag (global) x 4 B frags (LDS) per
// k-step -> 4 WMMAs; 16 k-steps -> 64 WMMAs per wave.
// ---------------------------------------------------------------------------
__device__ __forceinline__ void gemm_block_mainloop(
    const bf16* __restrict__ A, int mbase,
    const bf16* __restrict__ Bb /* batch's N-major panel base (nblk applied) */,
    bf16* bpanel /* LDS 64 x 128 */, int lane, int wv, v8f acc[4]) {
  int half = lane >> 4, n = lane & 15;
  for (int ks = 0; ks < DIMC; ks += 128) {
    if (wv == 0) {
      tdm_load_panel((unsigned)(size_t)bpanel, Bb + ks, 64, 128, DIMC);
      __builtin_amdgcn_s_wait_tensorcnt(0);
    }
    __syncthreads();
#pragma unroll
    for (int kk = 0; kk < 128; kk += 32) {
      v16bf a = load_a_frag(A, mbase, DIMC, ks + kk, lane);
#pragma unroll
      for (int j = 0; j < 4; ++j) {
        const bf16* p = bpanel + ((size_t)(j * 16 + n) << 7) + kk + (half << 4);
        v16bf bb;
#pragma unroll
        for (int e = 0; e < 16; ++e) bb[e] = p[e];
        acc[j] = wmma_bf16(a, bb, acc[j]);
      }
    }
    __syncthreads();   // panel consumed before next TDM overwrites it
  }
}

// ---------------------------------------------------------------------------
// QKV GEMM: per batch Out(1536 x 1024) = W(1536 x 512) * xn(512 x 1024) + b
// scatter into qT (b,h,q,d)*SCALE2, kbuf (b,h,kpos,d), vbuf (b,h,d,kpos)
// ---------------------------------------------------------------------------
__global__ void __launch_bounds__(128) qkv_gemm_kernel(
    const bf16* __restrict__ W, const bf16* __restrict__ X,
    const float* __restrict__ bias,
    bf16* __restrict__ qT, bf16* __restrict__ kbuf, bf16* __restrict__ vbuf) {
  __shared__ bf16 bpanel[64 * 128];
  int lane  = threadIdx.x & 31;
  int wv    = threadIdx.x >> 5;
  int b     = blockIdx.z;
  int nblk  = blockIdx.x * 64;
  int mbase = blockIdx.y * 64 + wv * 16;
  const bf16* Bb = X + ((size_t)b * HW + nblk) * DIMC;

  v8f acc[4] = {};
  gemm_block_mainloop(W, mbase, Bb, bpanel, lane, wv, acc);

  int half = lane >> 4, n = lane & 15;
#pragma unroll
  for (int j = 0; j < 4; ++j) {
    int hw = nblk + j * 16 + n;
#pragma unroll
    for (int r = 0; r < 8; ++r) {
      int o = mbase + r + (half << 3);
      float val = acc[j][r] + bias[o];
      int which = o >> 9;
      int head  = (o >> 6) & 7;
      int d     = o & 63;
      size_t bh = (size_t)b * HEADS + head;
      if (which == 0)
        qT[(bh * HW + hw) * HD + d] = (bf16)(val * SCALE2);
      else if (which == 1)
        kbuf[(bh * KVPAD + LCTX + hw) * HD + d] = (bf16)val;
      else
        vbuf[(bh * HD + d) * KVPAD + LCTX + hw] = (bf16)val;
    }
  }
}

// ---------------------------------------------------------------------------
// Output proj + residual: per batch Out(512 x 1024) = Wp(512x512) * AO + b + x
// ---------------------------------------------------------------------------
__global__ void __launch_bounds__(128) proj_gemm_kernel(
    const bf16* __restrict__ W, const bf16* __restrict__ X,
    const float* __restrict__ bias, const float* __restrict__ ident,
    float* __restrict__ out) {
  __shared__ bf16 bpanel[64 * 128];
  int lane  = threadIdx.x & 31;
  int wv    = threadIdx.x >> 5;
  int b     = blockIdx.z;
  int nblk  = blockIdx.x * 64;
  int mbase = blockIdx.y * 64 + wv * 16;
  const bf16* Bb = X + ((size_t)b * HW + nblk) * DIMC;

  v8f acc[4] = {};
  gemm_block_mainloop(W, mbase, Bb, bpanel, lane, wv, acc);

  int half = lane >> 4, n = lane & 15;
#pragma unroll
  for (int j = 0; j < 4; ++j) {
    int hw = nblk + j * 16 + n;
#pragma unroll
    for (int r = 0; r < 8; ++r) {
      int o = mbase + r + (half << 3);
      size_t idx = ((size_t)b * DIMC + o) * HW + hw;
      out[idx] = acc[j][r] + bias[o] + ident[idx];
    }
  }
}

// ---------------------------------------------------------------------------
// Context KV GEMM: per batch Out(1024 x 77) = W(1024 x 768) * ctx^T + b
// N edge (77) masked; one wave per 16x16 tile (tiny GEMM)
// ---------------------------------------------------------------------------
__global__ void ckv_gemm_kernel(const bf16* __restrict__ W, const bf16* __restrict__ C,
                                const float* __restrict__ bias,
                                bf16* __restrict__ kbuf, bf16* __restrict__ vbuf) {
  int lane  = threadIdx.x & 31;
  int wv    = threadIdx.x >> 5;
  int b     = blockIdx.z;
  int nbase = blockIdx.x * 16;
  int mbase = (blockIdx.y * 4 + wv) * 16;
  const bf16* Cb = C + (size_t)b * LCTX * CTXD;
  int half = lane >> 4, n = lane & 15;
  int l = nbase + n;

  v8f acc = {};
  for (int kk = 0; kk < CTXD; kk += 32) {
    v16bf a = load_a_frag(W, mbase, CTXD, kk, lane);
    v16bf bb;
    if (l < LCTX) {
      const bf16* p = Cb + (size_t)l * CTXD + kk + (half << 4);
#pragma unroll
      for (int e = 0; e < 16; ++e) bb[e] = p[e];
    } else {
#pragma unroll
      for (int e = 0; e < 16; ++e) bb[e] = (bf16)0.f;
    }
    acc = wmma_bf16(a, bb, acc);
  }
  if (l >= LCTX) return;
#pragma unroll
  for (int r = 0; r < 8; ++r) {
    int o = mbase + r + (half << 3);
    float val = acc[r] + bias[o];
    int kv = o >> 9, head = (o >> 6) & 7, d = o & 63;
    size_t bh = (size_t)b * HEADS + head;
    if (kv == 0) kbuf[(bh * KVPAD + l) * HD + d] = (bf16)val;
    else         vbuf[(bh * HD + d) * KVPAD + l] = (bf16)val;
  }
}

// ---------------------------------------------------------------------------
// Fused flash attention: one wave per 16-query tile of one (b, head).
// S(16x32) via 2x2 WMMAs, online softmax (16-lane shfl reductions),
// P re-fragmented through LDS, O(16x64) via 4 WMMAs per kv tile.
// ---------------------------------------------------------------------------
__global__ void attn_kernel(const bf16* __restrict__ qT, const bf16* __restrict__ kbuf,
                            const bf16* __restrict__ vbuf, bf16* __restrict__ aoT) {
  int lane  = threadIdx.x & 31;
  int wv    = threadIdx.x >> 5;
  int head  = blockIdx.y, b = blockIdx.z;
  int qbase = (blockIdx.x * 4 + wv) * 16;
  size_t bh = (size_t)b * HEADS + head;
  int half = lane >> 4, nn = lane & 15;

  __shared__ float pshm[4][16][32];

  v16bf aq0, aq1;
  {
    const bf16* p = qT + (bh * HW + qbase + nn) * HD + (half << 4);
#pragma unroll
    for (int e = 0; e < 8; ++e) { aq0[e] = p[e];      aq0[8 + e] = p[8 + e]; }
#pragma unroll
    for (int e = 0; e < 8; ++e) { aq1[e] = p[32 + e]; aq1[8 + e] = p[40 + e]; }
  }

  v8f o0 = {}, o1 = {}, o2 = {}, o3 = {};
  float mrow[8], lrow[8];
#pragma unroll
  for (int r = 0; r < 8; ++r) { mrow[r] = -3.0e38f; lrow[r] = 0.f; }

  const bf16* kb = kbuf + bh * (size_t)KVPAD * HD;  // (kpos, d)
  const bf16* vb = vbuf + bh * (size_t)HD * KVPAD;  // (d, kpos)

  for (int kt = 0; kt < KVPAD; kt += 32) {
    v8f s0 = {}, s1 = {};
    {
      const bf16* p0 = kb + (size_t)(kt + nn) * HD + (half << 4);
      const bf16* p1 = kb + (size_t)(kt + 16 + nn) * HD + (half << 4);
      v16bf b00, b01, b10, b11;
#pragma unroll
      for (int e = 0; e < 16; ++e) {
        b00[e] = p0[e]; b01[e] = p0[32 + e];
        b10[e] = p1[e]; b11[e] = p1[32 + e];
      }
      s0 = wmma_bf16(aq0, b00, s0); s0 = wmma_bf16(aq1, b01, s0);
      s1 = wmma_bf16(aq0, b10, s1); s1 = wmma_bf16(aq1, b11, s1);
    }

    bool v0 = (kt + nn) < KVLEN;
    bool v1 = (kt + 16 + nn) < KVLEN;

#pragma unroll
    for (int r = 0; r < 8; ++r) {
      float x0 = v0 ? s0[r] : -3.0e38f;
      float x1 = v1 ? s1[r] : -3.0e38f;
      float rmax = fmaxf(x0, x1);
#pragma unroll
      for (int off = 1; off < 16; off <<= 1)
        rmax = fmaxf(rmax, __shfl_xor(rmax, off, 32));
      float mnew = fmaxf(mrow[r], rmax);
      float corr = __expf(mrow[r] - mnew);
      float p0v  = __expf(x0 - mnew);
      float p1v  = __expf(x1 - mnew);
      float rsum = p0v + p1v;
#pragma unroll
      for (int off = 1; off < 16; off <<= 1)
        rsum += __shfl_xor(rsum, off, 32);
      lrow[r] = lrow[r] * corr + rsum;
      mrow[r] = mnew;
      o0[r] *= corr; o1[r] *= corr; o2[r] *= corr; o3[r] *= corr;
      pshm[wv][(half << 3) + r][nn]      = p0v;
      pshm[wv][(half << 3) + r][16 + nn] = p1v;
    }
    __syncthreads();

    v16bf pa;
#pragma unroll
    for (int e = 0; e < 16; ++e) {
      int kc = (half << 4) + e;
      pa[e] = (bf16)pshm[wv][nn][kc];
    }
    __syncthreads();

    const bf16* pvb = vb + kt + (half << 4);
    v16bf bv;
#pragma unroll
    for (int e = 0; e < 16; ++e) bv[e] = pvb[(size_t)(0 * 16 + nn) * KVPAD + e];
    o0 = wmma_bf16(pa, bv, o0);
#pragma unroll
    for (int e = 0; e < 16; ++e) bv[e] = pvb[(size_t)(1 * 16 + nn) * KVPAD + e];
    o1 = wmma_bf16(pa, bv, o1);
#pragma unroll
    for (int e = 0; e < 16; ++e) bv[e] = pvb[(size_t)(2 * 16 + nn) * KVPAD + e];
    o2 = wmma_bf16(pa, bv, o2);
#pragma unroll
    for (int e = 0; e < 16; ++e) bv[e] = pvb[(size_t)(3 * 16 + nn) * KVPAD + e];
    o3 = wmma_bf16(pa, bv, o3);
  }

  bf16* op = aoT + ((size_t)b * HW + qbase) * DIMC + head * HD;
#pragma unroll
  for (int r = 0; r < 8; ++r) {
    float inv = 1.0f / lrow[r];
    bf16* row = op + (size_t)((half << 3) + r) * DIMC;
    row[0 * 16 + nn] = (bf16)(o0[r] * inv);
    row[1 * 16 + nn] = (bf16)(o1[r] * inv);
    row[2 * 16 + nn] = (bf16)(o2[r] * inv);
    row[3 * 16 + nn] = (bf16)(o3[r] * inv);
  }
}

// ---------------------------------------------------------------------------
extern "C" void kernel_launch(void* const* d_in, const int* in_sizes, int n_in,
                              void* d_out, int out_size, void* d_ws, size_t ws_size,
                              hipStream_t stream) {
  (void)in_sizes; (void)n_in; (void)out_size; (void)ws_size;
  const float* x        = (const float*)d_in[0];
  const float* context  = (const float*)d_in[1];
  const float* gn_gamma = (const float*)d_in[2];
  const float* gn_beta  = (const float*)d_in[3];
  const float* qkv_w    = (const float*)d_in[4];
  const float* qkv_b    = (const float*)d_in[5];
  const float* ckv_w    = (const float*)d_in[6];
  const float* ckv_b    = (const float*)d_in[7];
  const float* proj_w   = (const float*)d_in[8];
  const float* proj_b   = (const float*)d_in[9];
  float* out = (float*)d_out;

  char* ws = (char*)d_ws;
  size_t off = 0;
  auto alloc = [&](size_t bytes) -> void* {
    void* p = ws + off;
    off += (bytes + 255) & ~(size_t)255;
    return p;
  };
  bf16* xnT   = (bf16*)alloc((size_t)BATCH * HW * DIMC * 2);
  bf16* wq    = (bf16*)alloc((size_t)3 * DIMC * DIMC * 2);
  bf16* wckv  = (bf16*)alloc((size_t)2 * DIMC * CTXD * 2);
  bf16* wproj = (bf16*)alloc((size_t)DIMC * DIMC * 2);
  bf16* ctxb  = (bf16*)alloc((size_t)BATCH * LCTX * CTXD * 2);
  bf16* qT    = (bf16*)alloc((size_t)BATCH * HEADS * HW * HD * 2);
  bf16* kbuf  = (bf16*)alloc((size_t)BATCH * HEADS * KVPAD * HD * 2);
  bf16* vbuf  = (bf16*)alloc((size_t)BATCH * HEADS * HD * KVPAD * 2);
  bf16* aoT   = (bf16*)alloc((size_t)BATCH * HW * DIMC * 2);

  (void)hipMemsetAsync(kbuf, 0, (size_t)BATCH * HEADS * KVPAD * HD * 2, stream);
  (void)hipMemsetAsync(vbuf, 0, (size_t)BATCH * HEADS * HD * KVPAD * 2, stream);

  auto conv = [&](const float* src, bf16* dst, int n) {
    f2bf_kernel<<<(n + 255) / 256, 256, 0, stream>>>(src, dst, n);
  };
  conv(qkv_w,  wq,    3 * DIMC * DIMC);
  conv(ckv_w,  wckv,  2 * DIMC * CTXD);
  conv(proj_w, wproj, DIMC * DIMC);
  conv(context, ctxb, BATCH * LCTX * CTXD);

  groupnorm_kernel<<<BATCH * NGRP, 256, 0, stream>>>(x, gn_gamma, gn_beta, xnT);

  qkv_gemm_kernel<<<dim3(HW / 64, (3 * DIMC) / 64, BATCH), 128, 0, stream>>>(
      wq, xnT, qkv_b, qT, kbuf, vbuf);

  ckv_gemm_kernel<<<dim3(5 /* ceil(77/16) */, (2 * DIMC) / 64, BATCH), 128, 0, stream>>>(
      wckv, ctxb, ckv_b, kbuf, vbuf);

  attn_kernel<<<dim3(HW / 64, HEADS, BATCH), 128, 0, stream>>>(qT, kbuf, vbuf, aoT);

  proj_gemm_kernel<<<dim3(HW / 64, DIMC / 64, BATCH), 128, 0, stream>>>(
      wproj, aoT, proj_b, x, out);
}
